// Attend_73203422593752
// MI455X (gfx1250) — matrix operand order repaired and checked by reading, not compiled, not run on hardware
//
#include <hip/hip_runtime.h>
#include <hip/hip_bf16.h>

typedef __attribute__((ext_vector_type(16))) __bf16 v16bf;
typedef __attribute__((ext_vector_type(8)))  float  v8f;
typedef unsigned int uint32x4 __attribute__((ext_vector_type(4)));
typedef int          int32x8  __attribute__((ext_vector_type(8)));
typedef int          int32x4  __attribute__((ext_vector_type(4)));

#define HID         256
#define ROWS_PER_WG 128   // 8 waves * 16 rows; == 2 full output rows (2*64 words)
#define THREADS     256
#define WBF_ELEMS   (HID * HID)          // 65536 bf16 = 128KB

__device__ __forceinline__ float fast_tanh(float v) {
    // tanh(v) = 1 - 2/(exp(2v)+1); exact limits at +-inf
    float e = __expf(2.0f * v);
    return 1.0f - 2.0f * __builtin_amdgcn_rcpf(e + 1.0f);
}

// ---------------------------------------------------------------------------
// Prep: W (f32, row-major) -> bf16 in B-fragment order:
//   ws[((kb*16 + j)*32 + lane)*16 + i] = W[j*16 + (lane&15)][kb*32 + (lane>>4)*16 + i]
// so each WMMA B-fragment is one contiguous 32B chunk per lane.
// ---------------------------------------------------------------------------
__global__ __launch_bounds__(256)
void prep_w_bf16(const float* __restrict__ Wm, __bf16* __restrict__ wsb)
{
    const int t    = blockIdx.x * 256 + threadIdx.x;   // 0..4095
    const int kb   = t >> 9;
    const int j    = (t >> 5) & 15;
    const int lane = t & 31;
    const int n    = j * 16 + (lane & 15);
    const int k0   = kb * 32 + (lane >> 4) * 16;
    const float* wp = Wm + (size_t)n * HID + k0;
    __bf16* dp = wsb + (size_t)t * 16;
    #pragma unroll
    for (int i = 0; i < 16; ++i) dp[i] = (__bf16)wp[i];
}

// ---------------------------------------------------------------------------
// Fused attention kernel
// ---------------------------------------------------------------------------
__global__ __launch_bounds__(THREADS)
void attend_fused(const float* __restrict__ x,
                  const __bf16* __restrict__ wbf,    // fragment-ordered bf16 W (d_ws)
                  const float* __restrict__ bias,
                  const float* __restrict__ ctx,
                  float* __restrict__ out)
{
    __shared__ __attribute__((aligned(32))) __bf16 lds_w[WBF_ELEMS]; // 128KB
    __shared__ float outacc[2 * HID];                                // 2 output rows

    const int tid  = threadIdx.x;
    const int lane = tid & 31;
    const int wv   = tid >> 5;          // wave 0..7
    const int half = lane >> 4;         // lane-half (fragment layout split)
    const int l16  = lane & 15;

    outacc[tid]           = 0.0f;
    outacc[tid + THREADS] = 0.0f;

    // ---- Stage the 128KB bf16 W tile into LDS via the Tensor Data Mover ----
#if defined(__gfx1250__) && __has_builtin(__builtin_amdgcn_tensor_load_to_lds)
    if (wv == 0) {
        // Tensor DMA descriptor (D#): linear 128KB copy as 16384 x 8B elements
        const unsigned long long ga = (unsigned long long)(const void*)wbf;
        uint32x4 g0;
        g0[0] = 1u;                                         // count=1, user mode
        g0[1] = (unsigned)(unsigned long long)(void*)lds_w; // lds_addr (LDS byte offset)
        g0[2] = (unsigned)(ga & 0xffffffffu);               // global_addr[31:0]
        g0[3] = (unsigned)((ga >> 32) & 0x01ffffffu)        // global_addr[56:32]
              | (2u << 30);                                 // type = 2 ("image")
        int32x8 g1;
        g1[0] = 0x30000;      // workgroup_mask=0, data_size=3 (8B)
        g1[1] = 0x40000000;   // tensor_dim0 = 16384 (low16 in bits 63:48)
        g1[2] = 0x00010000;   // tensor_dim0 hi = 0, tensor_dim1 = 1
        g1[3] = 0x40000000;   // tile_dim0 = 16384 (bits 127:112)
        g1[4] = 1;            // tile_dim1 = 1, tile_dim2 = 0
        g1[5] = 16384;        // tensor_dim0_stride (low 32)
        g1[6] = 0;            // stride hi, tensor_dim1_stride lo
        g1[7] = 0;
        int32x4 gz4 = {0, 0, 0, 0};
        int32x8 gz8 = {0, 0, 0, 0, 0, 0, 0, 0};
        __builtin_amdgcn_tensor_load_to_lds(g0, g1, gz4, gz4, gz8, 0);
        __builtin_amdgcn_s_wait_tensorcnt(0);
    }
#else
    {
        const uint4* src = (const uint4*)wbf;            // 8192 x 16B
        uint4* dst = (uint4*)lds_w;
        for (int i = tid; i < WBF_ELEMS / 8; i += THREADS) dst[i] = src[i];
    }
#endif
    __syncthreads();

    const int rowbase = blockIdx.x * ROWS_PER_WG + wv * 16;

    // ---------------- Phase 1: scores = tanh(x W^T + b) * ctx via WMMA ----------------
    v8f acc[16] = {};                                    // 16 N-tiles; full 16x256 in regs

    const float* xA = x + (size_t)(rowbase + l16) * HID; // A-frag: lane holds row M=l16
    const __bf16* lw = lds_w + lane * 16;                // per-lane fragment base

    for (int kb = 0; kb < 8; ++kb) {
        // A fragment 16x32 bf16: lanes 0-15 -> K 0..15; lanes 16-31 -> K 16..31
        v16bf a;
        {
            const float* p = xA + kb * 32 + half * 16;
            float4 f0 = *(const float4*)(p);
            float4 f1 = *(const float4*)(p + 4);
            float4 f2 = *(const float4*)(p + 8);
            float4 f3 = *(const float4*)(p + 12);
            a[0]=(__bf16)f0.x;  a[1]=(__bf16)f0.y;  a[2]=(__bf16)f0.z;  a[3]=(__bf16)f0.w;
            a[4]=(__bf16)f1.x;  a[5]=(__bf16)f1.y;  a[6]=(__bf16)f1.z;  a[7]=(__bf16)f1.w;
            a[8]=(__bf16)f2.x;  a[9]=(__bf16)f2.y;  a[10]=(__bf16)f2.z; a[11]=(__bf16)f2.w;
            a[12]=(__bf16)f3.x; a[13]=(__bf16)f3.y; a[14]=(__bf16)f3.z; a[15]=(__bf16)f3.w;
        }
        #pragma unroll
        for (int j = 0; j < 16; ++j) {
            // B fragment: one 32B ds_load per lane, pre-swizzled by prep kernel
            v16bf b = *(const v16bf*)(lw + kb * 8192 + j * 512);
            acc[j] = __builtin_amdgcn_wmma_f32_16x16x32_bf16(
                false, a, false, b, (short)0, acc[j], false, false);
        }
    }

    // bias + fast tanh + context  (C layout: element (r,lane) = row r+8*half, col j*16+l16)
    #pragma unroll
    for (int j = 0; j < 16; ++j) {
        const int col = j * 16 + l16;
        const float bv = bias[col];
        const float cv = ctx[col];
        #pragma unroll
        for (int r = 0; r < 8; ++r)
            acc[j][r] = fast_tanh(acc[j][r] + bv) * cv;
    }

    // ---------------- Phase 2: softmax over 256 cols, fully in registers ----------------
    float inv[8];
    #pragma unroll
    for (int r = 0; r < 8; ++r) {
        float pm = acc[0][r];
        #pragma unroll
        for (int j = 1; j < 16; ++j) pm = fmaxf(pm, acc[j][r]);
        pm = fmaxf(pm, __shfl_xor(pm, 1, 32));   // xor<=8 stays within the 16-lane half
        pm = fmaxf(pm, __shfl_xor(pm, 2, 32));
        pm = fmaxf(pm, __shfl_xor(pm, 4, 32));
        pm = fmaxf(pm, __shfl_xor(pm, 8, 32));
        float es = 0.0f;
        #pragma unroll
        for (int j = 0; j < 16; ++j) {
            float e = __expf(acc[j][r] - pm);
            acc[j][r] = e;
            es += e;
        }
        es += __shfl_xor(es, 1, 32);
        es += __shfl_xor(es, 2, 32);
        es += __shfl_xor(es, 4, 32);
        es += __shfl_xor(es, 8, 32);
        inv[r] = __builtin_amdgcn_rcpf(es);
    }

    // ---------------- Phase 3: attended = x*alpha, reduce 16 rows -> LDS ----------------
    const int orow = wv >> 2;   // 128 rows/WG = 2 output rows; wave wv feeds row wv/4
    #pragma unroll
    for (int j = 0; j < 16; ++j) {
        float po = 0.0f;
        #pragma unroll
        for (int r = 0; r < 8; ++r) {
            const int row = rowbase + r + 8 * half;
            const float xv = x[(size_t)row * HID + j * 16 + l16];
            po = fmaf(acc[j][r] * inv[r], xv, po);
        }
        po += __shfl_xor(po, 16, 32);            // combine lane-halves (rows 0-7 + 8-15)
        if (half == 0)
            atomicAdd(&outacc[orow * HID + j * 16 + l16], po);   // ds_add_f32
    }
    __syncthreads();

    // ---------------- write the 2 output rows ----------------
    float* ob = out + (size_t)blockIdx.x * (2 * HID);
    ob[tid]           = outacc[tid];
    ob[tid + THREADS] = outacc[tid + THREADS];
}

extern "C" void kernel_launch(void* const* d_in, const int* in_sizes, int n_in,
                              void* d_out, int out_size, void* d_ws, size_t ws_size,
                              hipStream_t stream) {
    const float* x   = (const float*)d_in[0];
    const float* Wm  = (const float*)d_in[1];
    const float* b   = (const float*)d_in[2];
    const float* ctx = (const float*)d_in[3];
    float* out = (float*)d_out;
    __bf16* wsb = (__bf16*)d_ws;                    // 128KB scratch for bf16 W

    prep_w_bf16<<<16, 256, 0, stream>>>(Wm, wsb);

    const int total_rows = in_sizes[0] / HID;       // N*WORDS = 131072
    const int blocks = total_rows / ROWS_PER_WG;    // 1024
    attend_fused<<<blocks, THREADS, 0, stream>>>(x, wsb, b, ctx, out);
}
